// FlashAttention_1623497638523
// MI455X (gfx1250) — matrix operand order repaired and checked by reading, not compiled, not run on hardware
//
#include <hip/hip_runtime.h>
#include <hip/hip_bf16.h>

typedef __attribute__((ext_vector_type(16))) _Float16 v16h;
typedef __attribute__((ext_vector_type(2)))  _Float16 v2h;
typedef __attribute__((ext_vector_type(8)))  float    v8f;

#define NH   16    // heads
#define SEQ  4096  // sequence length
#define DH   64    // head dim
#define BR   64    // query rows per workgroup (4 waves x 16)
#define BC   64    // keys per block (4 n-tiles, 2 k-chunks)
#define NWAVE 4

// Scores are computed in the log2 domain: Q is pre-scaled by (1/sqrt(D))*log2(e)
// so softmax exponentials are bare v_exp_f32 (exp2) with no ln2 fixup muls.

// One 64-key block: S = Q K^T, online softmax, O += P V.
// MASKED=true only for the single diagonal block (causal).
template<bool MASKED>
__device__ __forceinline__
void attn_block(int kb, int q0, int wave, int lmod, int lhal,
                const v16h (&aQ)[2], v8f (&acc)[4],
                float (&m_r)[8], float (&l_r)[8],
                _Float16 (&Ks)[BC][DH], _Float16 (&Vt)[DH][BC],
                _Float16 (&Ps)[16][BC])
{
    // ---- S = Q K^T : 4 n-tiles of 16 keys, contraction d=64 in 2 chunks ----
    v8f c[4] = {};
    #pragma unroll
    for (int kc = 0; kc < 2; ++kc) {
        #pragma unroll
        for (int n = 0; n < 4; ++n) {
            v16h b;
            #pragma unroll
            for (int e = 0; e < 16; ++e)
                b[e] = Ks[n * 16 + lmod][kc * 32 + ((e < 8) ? e : e + 8) + lhal * 8];
            c[n] = __builtin_amdgcn_wmma_f32_16x16x32_f16(false, aQ[kc], false, b,
                                                          (short)0, c[n], false, false);
        }
    }

    // ---- causal mask, applied destructively (diagonal block only) ----
    if (MASKED) {
        #pragma unroll
        for (int r = 0; r < 8; ++r) {
            int qg = q0 + wave * 16 + r + lhal * 8;
            #pragma unroll
            for (int n = 0; n < 4; ++n)
                if (kb * BC + n * 16 + lmod > qg) c[n][r] = -__builtin_inff();
        }
    }

    // ---- per-row local max over the 4 key-tiles ----
    float rmax[8];
    #pragma unroll
    for (int r = 0; r < 8; ++r)
        rmax[r] = fmaxf(fmaxf(c[0][r], c[1][r]), fmaxf(c[2][r], c[3][r]));

    // ---- packed-f16 butterfly max-reduce across the 16 lanes of each row ----
    // (row pairs share one dword -> half the bpermutes; f16-rounded max is a
    //  valid softmax stabilizer since it is used consistently below)
    #pragma unroll
    for (int j = 0; j < 4; ++j) {
        v2h pk; pk[0] = (_Float16)rmax[2 * j]; pk[1] = (_Float16)rmax[2 * j + 1];
        int bits = __builtin_bit_cast(int, pk);
        #pragma unroll
        for (int off = 1; off < 16; off <<= 1) {
            int o = __shfl_xor(bits, off, 16);
            v2h mx = __builtin_elementwise_max(__builtin_bit_cast(v2h, bits),
                                               __builtin_bit_cast(v2h, o));
            bits = __builtin_bit_cast(int, mx);
        }
        v2h mx = __builtin_bit_cast(v2h, bits);
        rmax[2 * j]     = (float)mx[0];
        rmax[2 * j + 1] = (float)mx[1];
    }

    // ---- online update + P store (C layout: row = r + lhal*8, col = lmod) ----
    #pragma unroll
    for (int r = 0; r < 8; ++r) {
        float mnew  = fmaxf(m_r[r], rmax[r]);
        float alpha = __builtin_amdgcn_exp2f(m_r[r] - mnew);
        m_r[r] = mnew;
        float p0 = __builtin_amdgcn_exp2f(c[0][r] - mnew);
        float p1 = __builtin_amdgcn_exp2f(c[1][r] - mnew);
        float p2 = __builtin_amdgcn_exp2f(c[2][r] - mnew);
        float p3 = __builtin_amdgcn_exp2f(c[3][r] - mnew);
        l_r[r] = l_r[r] * alpha + ((p0 + p1) + (p2 + p3));
        #pragma unroll
        for (int t = 0; t < 4; ++t) acc[t][r] *= alpha;
        int m = r + lhal * 8;
        Ps[m][ 0 + lmod] = (_Float16)p0;
        Ps[m][16 + lmod] = (_Float16)p1;
        Ps[m][32 + lmod] = (_Float16)p2;
        Ps[m][48 + lmod] = (_Float16)p3;
    }

    // ---- reload P as two A-matrix (16x32) fragments ----
    v16h aP[2];
    #pragma unroll
    for (int cc = 0; cc < 2; ++cc)
        #pragma unroll
        for (int e = 0; e < 16; ++e)
            aP[cc][e] = Ps[lmod][cc * 32 + ((e < 8) ? e : e + 8) + lhal * 8];

    // ---- O += P V : 4 d-tiles, contraction 64 keys in 2 chunks ----
    #pragma unroll
    for (int t = 0; t < 4; ++t) {
        #pragma unroll
        for (int cc = 0; cc < 2; ++cc) {
            v16h bV;
            #pragma unroll
            for (int e = 0; e < 16; ++e)
                bV[e] = Vt[t * 16 + lmod][cc * 32 + ((e < 8) ? e : e + 8) + lhal * 8];
            acc[t] = __builtin_amdgcn_wmma_f32_16x16x32_f16(false, aP[cc], false, bV,
                                                            (short)0, acc[t], false, false);
        }
    }
}

__device__ __forceinline__
void load_tile(int kb, int h, int tid,
               const float* __restrict__ k, const float* __restrict__ v,
               _Float16 (&Ks)[BC][DH], _Float16 (&Vt)[DH][BC])
{
    for (int i = tid; i < BC * DH; i += 128) {
        int key = i >> 6;
        int d   = i & 63;
        size_t g = ((size_t)(kb * BC + key) * NH + h) * DH + d;
        Ks[key][d] = (_Float16)k[g];
        Vt[d][key] = (_Float16)v[g];
    }
}

__device__ __forceinline__
void prefetch_tile(int kb, int h, int tid,
                   const float* __restrict__ k, const float* __restrict__ v)
{
    // 64 keys x 256B/row = 128 cachelines (128B); 128 threads -> 1 line each.
    int key  = tid >> 1;
    int doff = (tid & 1) * 32;
    size_t g = ((size_t)(kb * BC + key) * NH + h) * DH + doff;
    __builtin_prefetch(k + g, 0, 0);
    __builtin_prefetch(v + g, 0, 0);
}

__global__ __launch_bounds__(128)
void fa_fwd_wmma(const float* __restrict__ q, const float* __restrict__ k,
                 const float* __restrict__ v, const int* __restrict__ causal_p,
                 float* __restrict__ out)
{
    const int h    = blockIdx.y;
    const int q0   = blockIdx.x * BR;
    const int tid  = threadIdx.x;
    const int wave = tid >> 5;
    const int lane = tid & 31;
    const int lmod = lane & 15;
    const int lhal = lane >> 4;
    const int causal = *causal_p;

    __shared__ _Float16 Ks[BC][DH];
    __shared__ _Float16 Vt[DH][BC];
    __shared__ _Float16 Ps[NWAVE][16][BC];

    // ---- Q fragments (A 16x32 layout); fold 1/sqrt(D) * log2(e) into Q ----
    const float QSCALE = 0.125f * 1.4426950408889634f;
    const int qr = q0 + wave * 16 + lmod;
    const float* qrow = q + ((size_t)qr * NH + h) * DH;
    v16h aQ[2];
    #pragma unroll
    for (int kc = 0; kc < 2; ++kc)
        #pragma unroll
        for (int e = 0; e < 16; ++e) {
            int d = kc * 32 + ((e < 8) ? e : e + 8) + lhal * 8;
            aQ[kc][e] = (_Float16)(qrow[d] * QSCALE);
        }

    v8f acc[4] = {};
    float m_r[8], l_r[8];
    #pragma unroll
    for (int r = 0; r < 8; ++r) { m_r[r] = -__builtin_inff(); l_r[r] = 0.f; }

    // causal: full (unmasked) blocks strictly below the diagonal, then one
    // masked diagonal block (q0 is a multiple of BC=BR=64).
    const int nfull = causal ? (q0 / BC) : (SEQ / BC);

    for (int kb = 0; kb < nfull; ++kb) {
        __syncthreads();
        load_tile(kb, h, tid, k, v, Ks, Vt);
        if (kb + 1 < nfull || causal) prefetch_tile(kb + 1, h, tid, k, v);
        __syncthreads();
        attn_block<false>(kb, q0, wave, lmod, lhal, aQ, acc, m_r, l_r,
                          Ks, Vt, Ps[wave]);
    }
    if (causal) {
        const int kb = q0 / BC;
        __syncthreads();
        load_tile(kb, h, tid, k, v, Ks, Vt);
        __syncthreads();
        attn_block<true>(kb, q0, wave, lmod, lhal, aQ, acc, m_r, l_r,
                         Ks, Vt, Ps[wave]);
    }

    // ---- finalize: reduce row-sums, normalize, store fp32 ----
    #pragma unroll
    for (int r = 0; r < 8; ++r) {
        float l = l_r[r];
        #pragma unroll
        for (int off = 1; off < 16; off <<= 1)
            l += __shfl_xor(l, off, 16);
        float inv = 1.f / l;
        int qg = q0 + wave * 16 + r + lhal * 8;
        float* orow = out + ((size_t)qg * NH + h) * DH;
        #pragma unroll
        for (int t = 0; t < 4; ++t)
            orow[t * 16 + lmod] = acc[t][r] * inv;
    }
}

extern "C" void kernel_launch(void* const* d_in, const int* in_sizes, int n_in,
                              void* d_out, int out_size, void* d_ws, size_t ws_size,
                              hipStream_t stream) {
    const float* q      = (const float*)d_in[0];
    const float* k      = (const float*)d_in[1];
    const float* v      = (const float*)d_in[2];
    const int*   causal = (const int*)d_in[3];
    float*       out    = (float*)d_out;

    dim3 grid(SEQ / BR, NH, 1);
    fa_fwd_wmma<<<grid, 128, 0, stream>>>(q, k, v, causal, out);
}
